// iKT3_67860483276897
// MI455X (gfx1250) — compile-verified, matrix operand order checked by reading, not run on hardware
//
#include <hip/hip_runtime.h>

#define B_   16
#define L_   1024
#define D_   256
#define H_   4
#define NB_  8
#define DFF_ 1536
#define DK_  64
#define M_   (B_ * L_)

typedef __attribute__((ext_vector_type(16))) __bf16 v16bf;
typedef __attribute__((ext_vector_type(8)))  __bf16 v8bf;
typedef __attribute__((ext_vector_type(8)))  float  v8f;

// Types matching the async-to-LDS builtin's parameters:
// param0: int __attribute__((vector_size(16))) addrspace(1)*  (global)
// param1: same vector type in LDS addrspace(3)
typedef int v4i_gcc __attribute__((vector_size(16)));
typedef __attribute__((address_space(1))) v4i_gcc* as1_v4ip;
typedef __attribute__((address_space(3))) v4i_gcc* as3_v4ip;

#if __has_builtin(__builtin_amdgcn_global_load_async_to_lds_b128)
#define USE_ASYNC_LDS 1
#else
#define USE_ASYNC_LDS 0
#endif

__device__ __forceinline__ v8f wmma_bf16(v16bf a, v16bf b, v8f c) {
  return __builtin_amdgcn_wmma_f32_16x16x32_bf16(false, a, false, b, (short)0, c,
                                                 false, false);
}

// 16-byte global -> LDS copy: async DMA (ASYNCcnt) when available.
__device__ __forceinline__ void copy16(const __bf16* g, __bf16* l) {
#if USE_ASYNC_LDS
  __builtin_amdgcn_global_load_async_to_lds_b128(
      (as1_v4ip)(void*)const_cast<__bf16*>(g), (as3_v4ip)(void*)l, 0, 0);
#else
  *(v8bf*)l = *(const v8bf*)g;
#endif
}

__device__ __forceinline__ void wait_async_copies() {
#if USE_ASYNC_LDS
#if __has_builtin(__builtin_amdgcn_s_wait_asynccnt)
  __builtin_amdgcn_s_wait_asynccnt(0);
#else
  asm volatile("s_wait_asynccnt 0" ::: "memory");
#endif
#endif
}

// Load a 16x32 (16-bit) WMMA A/B fragment from LDS laid out row-major with
// leading dimension `ld`. Lane l%16 -> row, (l/16)*8 -> K sub-block base.
__device__ __forceinline__ v16bf frag_lds(const __bf16* base, int ld, int lane) {
  const int rr = lane & 15;
  const int kb = (lane >> 4) << 3;
  const __bf16* p = base + rr * ld + kb;
  v16bf f;
#pragma unroll
  for (int i = 0; i < 8; ++i) f[i] = p[i];
#pragma unroll
  for (int i = 0; i < 8; ++i) f[8 + i] = p[16 + i];
  return f;
}

// ---------------------------------------------------------------------------
// fp32 [K,N] -> bf16 transposed [N,K] (weights converted on-device per launch;
// transposed layout makes every GEMM LDS tile a contiguous 16B async copy).
// ---------------------------------------------------------------------------
__global__ __launch_bounds__(256) void cvt_t_kernel(const float* __restrict__ in,
                                                    __bf16* __restrict__ out,
                                                    int K, int N) {
  int i = blockIdx.x * 256 + threadIdx.x;
  if (i < K * N) {
    int k = i / N, n = i - k * N;
    out[(size_t)n * K + k] = (__bf16)in[i];
  }
}

// ---------------------------------------------------------------------------
// Embedding: h = ctx_emb[q*2+r] + pos_emb[l]   (fp32 master + bf16 mirror)
// ---------------------------------------------------------------------------
__global__ __launch_bounds__(256) void embed_kernel(const int* __restrict__ q,
                                                    const int* __restrict__ r,
                                                    const float* __restrict__ ce,
                                                    const float* __restrict__ pe,
                                                    float* __restrict__ hm,
                                                    __bf16* __restrict__ hb) {
  int row = blockIdx.x;
  int t = threadIdx.x;
  int l = row & (L_ - 1);
  int tok = q[row] * 2 + r[row];
  float v = ce[(size_t)tok * D_ + t] + pe[(size_t)l * D_ + t];
  hm[(size_t)row * D_ + t] = v;
  hb[(size_t)row * D_ + t] = (__bf16)v;
}

// ---------------------------------------------------------------------------
// WMMA GEMM: C[M,N] = A[M,K](bf16) @ W(bf16, pre-transposed [N,K]) + bias.
// Block tile 128x128, K-step 32, 8 waves -> 4x2 wave grid of 32x64 subtiles
// (8 WMMAs per wave per K-step against 6 LDS fragment loads).
// Double-buffered LDS, async global->LDS DMA overlapped with WMMA.
// ---------------------------------------------------------------------------
#define BM 128
#define BN 128
#define BK 32
#define LDT (BK + 8)

__global__ __launch_bounds__(256)
void gemm_bf16_kernel(const __bf16* __restrict__ A, const __bf16* __restrict__ Wt,
                      const float* __restrict__ bias, float* __restrict__ outF,
                      __bf16* __restrict__ outB, int M, int N, int K, int relu) {
  __shared__ __align__(16) __bf16 As[2][BM][LDT];
  __shared__ __align__(16) __bf16 Bs[2][BN][LDT];  // [n][k]
  const int tid = threadIdx.x;
  const int lane = tid & 31;
  const int wid = tid >> 5;
  const int wm = wid & 3;   // 4 x 32 rows
  const int wn = wid >> 2;  // 2 x 64 cols
  const int bm = blockIdx.x * BM;
  const int bn = blockIdx.y * BN;

  v8f acc[2][4];
#pragma unroll
  for (int a0 = 0; a0 < 2; ++a0)
#pragma unroll
    for (int b0 = 0; b0 < 4; ++b0)
#pragma unroll
      for (int i = 0; i < 8; ++i) acc[a0][b0][i] = 0.0f;

  // 16B-chunk mapping: row = tid/4 (0..63), col = (tid%4)*8 elements.
  const int crow = tid >> 2;
  const int ccol = (tid & 3) * 8;

  auto issue = [&](int k0, int bi) {
    copy16(A + (size_t)(bm + crow) * K + k0 + ccol,       &As[bi][crow][ccol]);
    copy16(A + (size_t)(bm + crow + 64) * K + k0 + ccol,  &As[bi][crow + 64][ccol]);
    copy16(Wt + (size_t)(bn + crow) * K + k0 + ccol,      &Bs[bi][crow][ccol]);
    copy16(Wt + (size_t)(bn + crow + 64) * K + k0 + ccol, &Bs[bi][crow + 64][ccol]);
  };

  const int nt = K / BK;
  issue(0, 0);
  for (int t = 0; t < nt; ++t) {
    wait_async_copies();   // this wave's DMA for tile t has landed
    __syncthreads();       // everyone's tile t landed; tile t-1 consumers done
    if (t + 1 < nt) issue((t + 1) * BK, (t + 1) & 1);
    const int bi = t & 1;

    v16bf af[2], bfr[4];
#pragma unroll
    for (int fm = 0; fm < 2; ++fm)
      af[fm] = frag_lds(&As[bi][wm * 32 + fm * 16][0], LDT, lane);
#pragma unroll
    for (int fn = 0; fn < 4; ++fn)
      bfr[fn] = frag_lds(&Bs[bi][wn * 64 + fn * 16][0], LDT, lane);
#pragma unroll
    for (int fm = 0; fm < 2; ++fm)
#pragma unroll
      for (int fn = 0; fn < 4; ++fn)
        acc[fm][fn] = wmma_bf16(af[fm], bfr[fn], acc[fm][fn]);
  }

  const int hrow = (lane >> 4) << 3;
  const int ln16 = lane & 15;
#pragma unroll
  for (int fm = 0; fm < 2; ++fm)
#pragma unroll
    for (int fn = 0; fn < 4; ++fn) {
      int col = bn + wn * 64 + fn * 16 + ln16;
      float bv = bias ? bias[col] : 0.0f;
#pragma unroll
      for (int j = 0; j < 8; ++j) {
        int row_ = bm + wm * 32 + fm * 16 + hrow + j;
        float v = acc[fm][fn][j] + bv;
        if (relu) v = fmaxf(v, 0.0f);
        if (outF) outF[(size_t)row_ * N + col] = v;
        if (outB) outB[(size_t)row_ * N + col] = (__bf16)v;
      }
    }
}

// ---------------------------------------------------------------------------
// Causal flash attention, one wave per 16 query rows, WMMA for QK^T and PV.
// Q/K/V are [B,L,D] bf16 with head h at columns [h*64, h*64+64).
// ---------------------------------------------------------------------------
__global__ __launch_bounds__(256)
void attn_kernel(const __bf16* __restrict__ Qb, const __bf16* __restrict__ Kb,
                 const __bf16* __restrict__ Vb, __bf16* __restrict__ Ob) {
  __shared__ __bf16 Pl[8][16][40];  // per-wave P staging (C-layout -> A-layout)
  const int lane = threadIdx.x & 31;
  const int wid = threadIdx.x >> 5;
  const int bh = blockIdx.x;
  const int b = bh >> 2;  // H_ == 4
  const int h = bh & 3;
  const int qbase = blockIdx.y * 128 + wid * 16;
  const int lm = lane & 15;
  const int kb = (lane >> 4) << 3;
  const int hrow = kb;

  const __bf16* Qp = Qb + ((size_t)b * L_) * D_ + h * DK_;
  const __bf16* Kp = Kb + ((size_t)b * L_) * D_ + h * DK_;
  const __bf16* Vp = Vb + ((size_t)b * L_) * D_ + h * DK_;

  v16bf qf[2];
  {
    const __bf16* qr = Qp + (size_t)(qbase + lm) * D_;
#pragma unroll
    for (int s = 0; s < 2; ++s) {
#pragma unroll
      for (int i = 0; i < 8; ++i) {
        qf[s][i] = qr[s * 32 + kb + i];
        qf[s][8 + i] = qr[s * 32 + kb + 16 + i];
      }
    }
  }

  float mi[8], li[8];
  v8f ov[4];
#pragma unroll
  for (int j = 0; j < 8; ++j) { mi[j] = -1e30f; li[j] = 0.0f; }
#pragma unroll
  for (int n = 0; n < 4; ++n)
#pragma unroll
    for (int j = 0; j < 8; ++j) ov[n][j] = 0.0f;

  __bf16(*P)[40] = Pl[wid];
  const int kend = qbase + 16;
  const int ntiles = (kend + 31) >> 5;

  for (int t = 0; t < ntiles; ++t) {
    const int kt = t * 32;
    v8f s0, s1;
#pragma unroll
    for (int j = 0; j < 8; ++j) { s0[j] = 0.0f; s1[j] = 0.0f; }

#pragma unroll
    for (int half = 0; half < 2; ++half) {
      const __bf16* kr = Kp + (size_t)(kt + half * 16 + lm) * D_;
#pragma unroll
      for (int s = 0; s < 2; ++s) {
        v16bf kf;
#pragma unroll
        for (int i = 0; i < 8; ++i) {
          kf[i] = kr[s * 32 + kb + i];
          kf[8 + i] = kr[s * 32 + kb + 16 + i];
        }
        if (half == 0) s0 = wmma_bf16(qf[s], kf, s0);
        else           s1 = wmma_bf16(qf[s], kf, s1);
      }
    }

    float pn[2][8];
#pragma unroll
    for (int j = 0; j < 8; ++j) {
      int qi = qbase + hrow + j;
      pn[0][j] = (kt + lm <= qi)      ? s0[j] * 0.125f : -1e30f;
      pn[1][j] = (kt + 16 + lm <= qi) ? s1[j] * 0.125f : -1e30f;
    }
#pragma unroll
    for (int j = 0; j < 8; ++j) {
      float v = fmaxf(pn[0][j], pn[1][j]);
      v = fmaxf(v, __shfl_xor(v, 1));
      v = fmaxf(v, __shfl_xor(v, 2));
      v = fmaxf(v, __shfl_xor(v, 4));
      v = fmaxf(v, __shfl_xor(v, 8));
      float mn = fmaxf(mi[j], v);
      float corr = __expf(mi[j] - mn);
      float p0 = __expf(pn[0][j] - mn);
      float p1 = __expf(pn[1][j] - mn);
      P[hrow + j][lm] = (__bf16)p0;
      P[hrow + j][16 + lm] = (__bf16)p1;
      float rs = p0 + p1;
      rs += __shfl_xor(rs, 1);
      rs += __shfl_xor(rs, 2);
      rs += __shfl_xor(rs, 4);
      rs += __shfl_xor(rs, 8);
      li[j] = li[j] * corr + rs;
      mi[j] = mn;
#pragma unroll
      for (int n = 0; n < 4; ++n) ov[n][j] *= corr;
    }
    // wave-local LDS visibility (no block barrier: waves have different trip counts)
    asm volatile("s_wait_dscnt 0" ::: "memory");

    v16bf pf;
#pragma unroll
    for (int i = 0; i < 8; ++i) {
      pf[i] = P[lm][kb + i];
      pf[8 + i] = P[lm][kb + 16 + i];
    }

#pragma unroll
    for (int n = 0; n < 4; ++n) {
      v16bf vf;
#pragma unroll
      for (int i = 0; i < 8; ++i) {
        vf[i] = Vp[(size_t)(kt + kb + i) * D_ + n * 16 + lm];
        vf[8 + i] = Vp[(size_t)(kt + kb + 16 + i) * D_ + n * 16 + lm];
      }
      ov[n] = wmma_bf16(pf, vf, ov[n]);
    }
  }

#pragma unroll
  for (int n = 0; n < 4; ++n) {
#pragma unroll
    for (int j = 0; j < 8; ++j) {
      int q_ = qbase + hrow + j;
      float inv = 1.0f / li[j];
      Ob[((size_t)b * L_ + q_) * D_ + h * DK_ + n * 16 + lm] = (__bf16)(ov[n][j] * inv);
    }
  }
}

// ---------------------------------------------------------------------------
// h = LayerNorm(h + delta) * g + b ; writes fp32 master and bf16 mirror.
// ---------------------------------------------------------------------------
__global__ __launch_bounds__(256)
void resid_ln_kernel(float* __restrict__ hm, const float* __restrict__ delta,
                     const float* __restrict__ g, const float* __restrict__ bt,
                     __bf16* __restrict__ hb) {
  __shared__ float red[256];
  int row = blockIdx.x, t = threadIdx.x;
  size_t idx = (size_t)row * D_ + t;
  float x = hm[idx] + delta[idx];
  red[t] = x;
  __syncthreads();
  for (int s = 128; s > 0; s >>= 1) {
    if (t < s) red[t] += red[t + s];
    __syncthreads();
  }
  float mu = red[0] * (1.0f / D_);
  __syncthreads();
  float d = x - mu;
  red[t] = d * d;
  __syncthreads();
  for (int s = 128; s > 0; s >>= 1) {
    if (t < s) red[t] += red[t + s];
    __syncthreads();
  }
  float var = red[0] * (1.0f / D_);
  float y = d * rsqrtf(var + 1e-5f) * g[t] + bt[t];
  hm[idx] = y;
  hb[idx] = (__bf16)y;
}

// ---------------------------------------------------------------------------
// concat[row] = [ h_bf16[row], bf16(skill_tab[qry[row]]) ]
// ---------------------------------------------------------------------------
__global__ __launch_bounds__(256)
void concat_kernel(const __bf16* __restrict__ hb, const float* __restrict__ skill,
                   const int* __restrict__ qry, __bf16* __restrict__ cc) {
  int row = blockIdx.x, t = threadIdx.x;
  cc[(size_t)row * (2 * D_) + t] = hb[(size_t)row * D_ + t];
  cc[(size_t)row * (2 * D_) + D_ + t] = (__bf16)skill[(size_t)qry[row] * D_ + t];
}

// ---------------------------------------------------------------------------
// theta = ffa[row,:] . Aw2 + Ab2 ; out = sigmoid(scale*theta - beta[qry[row]])
// ---------------------------------------------------------------------------
__global__ __launch_bounds__(256)
void head_kernel(const __bf16* __restrict__ ffa, const float* __restrict__ Aw2,
                 const float* __restrict__ Ab2, const float* __restrict__ scale,
                 const float* __restrict__ beta, const int* __restrict__ qry,
                 float* __restrict__ out) {
  __shared__ float red[256];
  int row = blockIdx.x, t = threadIdx.x;
  float s = 0.0f;
  for (int j = t; j < DFF_; j += 256) s += (float)ffa[(size_t)row * DFF_ + j] * Aw2[j];
  red[t] = s;
  __syncthreads();
  for (int k = 128; k > 0; k >>= 1) {
    if (t < k) red[t] += red[t + k];
    __syncthreads();
  }
  if (t == 0) {
    float theta = red[0] + Ab2[0];
    float z = scale[0] * theta - beta[qry[row]];
    out[row] = 1.0f / (1.0f + __expf(-z));
  }
}

// ---------------------------------------------------------------------------
extern "C" void kernel_launch(void* const* d_in, const int* in_sizes, int n_in,
                              void* d_out, int out_size, void* d_ws, size_t ws_size,
                              hipStream_t stream) {
  (void)in_sizes; (void)n_in; (void)out_size; (void)ws_size;
  const int* qi   = (const int*)d_in[0];
  const int* ri   = (const int*)d_in[1];
  const int* qry  = (const int*)d_in[2];
  const float* ctx = (const float*)d_in[3];
  const float* pos = (const float*)d_in[4];
  const float* skl = (const float*)d_in[5];
  const float* Wq = (const float*)d_in[6];  const float* bq = (const float*)d_in[7];
  const float* Wk = (const float*)d_in[8];  const float* bk = (const float*)d_in[9];
  const float* Wv = (const float*)d_in[10]; const float* bv = (const float*)d_in[11];
  const float* Wo = (const float*)d_in[12]; const float* bo = (const float*)d_in[13];
  const float* ln1g = (const float*)d_in[14]; const float* ln1b = (const float*)d_in[15];
  const float* W1 = (const float*)d_in[16]; const float* b1 = (const float*)d_in[17];
  const float* W2 = (const float*)d_in[18]; const float* b2 = (const float*)d_in[19];
  const float* ln2g = (const float*)d_in[20]; const float* ln2b = (const float*)d_in[21];
  const float* Aw1 = (const float*)d_in[22]; const float* Ab1 = (const float*)d_in[23];
  const float* Aw2 = (const float*)d_in[24]; const float* Ab2 = (const float*)d_in[25];
  const float* scl = (const float*)d_in[26]; const float* beta = (const float*)d_in[27];
  float* out = (float*)d_out;

  char* ws = (char*)d_ws;
  size_t off = 0;
  float* hm = (float*)(ws + off);        off += (size_t)M_ * D_ * 4;     // 16 MB
  float* tmp = (float*)(ws + off);       off += (size_t)M_ * D_ * 4;     // 16 MB
  __bf16* hb = (__bf16*)(ws + off);      off += (size_t)M_ * D_ * 2;     // 8 MB
  __bf16* Qb = (__bf16*)(ws + off);      off += (size_t)M_ * D_ * 2;
  __bf16* Kbuf = (__bf16*)(ws + off);    off += (size_t)M_ * D_ * 2;
  __bf16* Vbuf = (__bf16*)(ws + off);    off += (size_t)M_ * D_ * 2;
  __bf16* attnb = (__bf16*)(ws + off);   off += (size_t)M_ * D_ * 2;
  __bf16* ff1 = (__bf16*)(ws + off);     off += (size_t)M_ * DFF_ * 2;   // 48 MB
  __bf16* catb = (__bf16*)(ws + off);    off += (size_t)M_ * 2 * D_ * 2; // 16 MB
  __bf16* wbuf = (__bf16*)(ws + off);    off += (size_t)2 * D_ * DFF_ * 2;

  embed_kernel<<<M_, 256, 0, stream>>>(qi, ri, ctx, pos, hm, hb);

  const dim3 gD(M_ / BM, D_ / BN);     // N = 256 -> 2 col-tiles
  const dim3 gF(M_ / BM, DFF_ / BN);   // N = 1536 -> 12 col-tiles

  for (int i = 0; i < NB_; ++i) {
    size_t oDD = (size_t)i * D_ * D_;
    size_t oD = (size_t)i * D_;
    size_t oDF = (size_t)i * D_ * DFF_;

    cvt_t_kernel<<<(D_ * D_) / 256, 256, 0, stream>>>(Wq + oDD, wbuf, D_, D_);
    gemm_bf16_kernel<<<gD, 256, 0, stream>>>(hb, wbuf, bq + oD, nullptr, Qb, M_, D_, D_, 0);
    cvt_t_kernel<<<(D_ * D_) / 256, 256, 0, stream>>>(Wk + oDD, wbuf, D_, D_);
    gemm_bf16_kernel<<<gD, 256, 0, stream>>>(hb, wbuf, bk + oD, nullptr, Kbuf, M_, D_, D_, 0);
    cvt_t_kernel<<<(D_ * D_) / 256, 256, 0, stream>>>(Wv + oDD, wbuf, D_, D_);
    gemm_bf16_kernel<<<gD, 256, 0, stream>>>(hb, wbuf, bv + oD, nullptr, Vbuf, M_, D_, D_, 0);

    attn_kernel<<<dim3(B_ * H_, L_ / 128), 256, 0, stream>>>(Qb, Kbuf, Vbuf, attnb);

    cvt_t_kernel<<<(D_ * D_) / 256, 256, 0, stream>>>(Wo + oDD, wbuf, D_, D_);
    gemm_bf16_kernel<<<gD, 256, 0, stream>>>(attnb, wbuf, bo + oD, tmp, nullptr, M_, D_, D_, 0);
    resid_ln_kernel<<<M_, 256, 0, stream>>>(hm, tmp, ln1g + oD, ln1b + oD, hb);

    cvt_t_kernel<<<(D_ * DFF_) / 256, 256, 0, stream>>>(W1 + oDF, wbuf, D_, DFF_);
    gemm_bf16_kernel<<<gF, 256, 0, stream>>>(hb, wbuf, b1 + (size_t)i * DFF_, nullptr, ff1,
                                             M_, DFF_, D_, 1);
    cvt_t_kernel<<<(DFF_ * D_) / 256, 256, 0, stream>>>(W2 + oDF, wbuf, DFF_, D_);
    gemm_bf16_kernel<<<gD, 256, 0, stream>>>(ff1, wbuf, b2 + oD, tmp, nullptr, M_, D_, DFF_, 0);
    resid_ln_kernel<<<M_, 256, 0, stream>>>(hm, tmp, ln2g + oD, ln2b + oD, hb);
  }

  concat_kernel<<<M_, 256, 0, stream>>>(hb, skl, qry, catb);
  cvt_t_kernel<<<(2 * D_ * DFF_) / 256, 256, 0, stream>>>(Aw1, wbuf, 2 * D_, DFF_);
  gemm_bf16_kernel<<<gF, 256, 0, stream>>>(catb, wbuf, Ab1, nullptr, ff1, M_, DFF_, 2 * D_, 1);
  head_kernel<<<M_, 256, 0, stream>>>(ff1, Aw2, Ab2, scl, beta, qry, out);
}
